// Self_attention_layer_67611375173952
// MI455X (gfx1250) — matrix operand order repaired and checked by reading, not compiled, not run on hardware
//
#include <hip/hip_runtime.h>
#include <hip/hip_bf16.h>

// ---------------- types ----------------
typedef __attribute__((ext_vector_type(16))) _Float16 v16h;
typedef __attribute__((ext_vector_type(8)))  _Float16 h8;
typedef __attribute__((ext_vector_type(4)))  _Float16 h4;
typedef __attribute__((ext_vector_type(8)))  float    v8f;
typedef __attribute__((ext_vector_type(4)))  float    f4;

// ---------------- problem constants ----------------
#define S_LEN  2048
#define D_DIM  512
#define BATCH  8
#define MQ     64      // query rows per workgroup (4 blocks of 16)
#define KC     64      // keys per chunk (4 blocks of 16)
#define NTHREADS 512   // 16 waves

// LDS pitches (elements) — padded for bank spread, 16B-aligned rows
#define QP   520      // Q tile pitch (halfs), 64 rows
#define KP   520      // K tile pitch (halfs), 64 rows
#define VTP  72       // V^T pitch (halfs), 512 rows (d-major)
#define SCP  65       // score pitch (floats), 64 rows
#define PP   72       // P pitch (halfs), 64 rows

#define SMEM_BYTES (MQ*QP*2 + KC*KP*2 + D_DIM*VTP*2 + MQ*SCP*4 + MQ*PP*2 + 3*MQ*4)

union V16u { v16h v; h8 h2[2]; h4 h4s[4]; };

// A-operand (f16 16x32, M x K) from row-major LDS tile.
// lane -> row M = lane&15; half-wave selects K base 0/8;
// VGPR0-3 hold K=kb..kb+7, VGPR4-7 hold K=kb+16..kb+23.
__device__ __forceinline__ v16h lds_loadA(const _Float16* base, int pitch, int k0, int lane) {
    int m  = lane & 15;
    int kb = (lane >> 4) * 8;
    const _Float16* p = base + m * pitch + k0 + kb;
    V16u u;
    u.h2[0] = *(const h8*)(p);
    u.h2[1] = *(const h8*)(p + 16);
    return u.v;
}

// B-operand (f16 32x16, K x N) from LDS where each "N" entity is a row of
// contiguous K values (K-tile rows / V^T rows). lane -> N = lane&15;
// half-wave selects K base 0/8; elements 0-7 hold K=kb..kb+7, 8-15 hold K=kb+16..kb+23.
__device__ __forceinline__ v16h lds_loadB(const _Float16* base, int pitch, int k0, int lane) {
    int n  = lane & 15;
    int kb = (lane >> 4) * 8;
    const _Float16* p = base + n * pitch + k0 + kb;
    V16u u;
    u.h2[0] = *(const h8*)(p);
    u.h2[1] = *(const h8*)(p + 16);
    return u.v;
}

__device__ __forceinline__ v8f wmma_f16(v16h a, v16h b, v8f c) {
    return __builtin_amdgcn_wmma_f32_16x16x32_f16(
        /*neg_a=*/false, a, /*neg_b=*/false, b,
        /*c_mod=*/(short)0, c, /*reuse_a=*/false, /*reuse_b=*/false);
}

extern "C" __global__ void __launch_bounds__(NTHREADS, 1)
flash_attn_cauchy_kernel(const float* __restrict__ Q,
                         const float* __restrict__ K,
                         const float* __restrict__ V,
                         float* __restrict__ O)
{
    extern __shared__ char smem_raw[];
    _Float16* Qs = (_Float16*)smem_raw;                 // [64][QP]  f16 query tile
    _Float16* Ks = Qs + MQ * QP;                        // [64][KP]  f16 key chunk
    _Float16* Vt = Ks + KC * KP;                        // [512][VTP] f16 V^T chunk
    float*    Sc = (float*)(Vt + D_DIM * VTP);          // [64][SCP] f32 scores
    _Float16* Pb = (_Float16*)(Sc + MQ * SCP);          // [64][PP]  f16 probs
    float*    smM = (float*)(Pb + MQ * PP);             // [64] running max
    float*    smL = smM + MQ;                           // [64] running sum
    float*    smA = smL + MQ;                           // [64] rescale alpha

    const int tid   = threadIdx.x;
    const int lane  = tid & 31;
    const int w     = tid >> 5;          // wave id 0..15
    const int hi    = lane >> 4;         // half-wave
    const int bidx  = blockIdx.y;        // batch
    const int qbase = blockIdx.x * MQ;

    const float* Qg = Q + ((size_t)bidx * S_LEN + qbase) * D_DIM;
    const float* Kg = K + (size_t)bidx * S_LEN * D_DIM;
    const float* Vg = V + (size_t)bidx * S_LEN * D_DIM;

    // ---- stage Q tile to LDS as f16 (coalesced f32 reads) ----
    #pragma unroll
    for (int it = 0; it < (MQ * D_DIM) / (NTHREADS * 4); ++it) {
        int flat = (it * NTHREADS + tid) * 4;
        int m = flat >> 9;             // /512
        int d = flat & (D_DIM - 1);
        f4 qv = *(const f4*)(Qg + flat);
        *(h4*)(Qs + m * QP + d) = __builtin_convertvector(qv, h4);
    }
    if (tid < MQ) { smM[tid] = -3.0e38f; smL[tid] = 0.0f; }
    __syncthreads();

    const int wm = w & 3;        // query 16-row block (both phases)
    const int wk = w >> 2;       // score phase: key 16-col block (0..3)
    const int wd = w >> 2;       // PV phase: D quarter (0..3), 128 cols each

    v8f acc[8];
    #pragma unroll
    for (int t = 0; t < 8; ++t) acc[t] = (v8f){0.f,0.f,0.f,0.f,0.f,0.f,0.f,0.f};

    const float scale = 0.022097086912079608f;   // 1/sqrt(2048)
    const int nchunks = (qbase + MQ - 1) / KC + 1;

    for (int ch = 0; ch < nchunks; ++ch) {
        const int kb0 = ch * KC;

        // ---- stage K chunk (row-major) and V^T chunk to LDS as f16 ----
        #pragma unroll
        for (int it = 0; it < (KC * D_DIM) / (NTHREADS * 4); ++it) {
            int flat = (it * NTHREADS + tid) * 4;
            int key = flat >> 9;           // /512
            int d   = flat & (D_DIM - 1);
            f4 kv = *(const f4*)(Kg + (size_t)kb0 * D_DIM + flat);
            *(h4*)(Ks + key * KP + d) = __builtin_convertvector(kv, h4);
            f4 vv = *(const f4*)(Vg + (size_t)kb0 * D_DIM + flat);
            #pragma unroll
            for (int j = 0; j < 4; ++j)
                Vt[(d + j) * VTP + key] = (_Float16)vv[j];
        }

        // prefetch next chunk's K/V into the cache hierarchy
        if (ch + 1 < nchunks) {
            const float* nk = Kg + (size_t)(kb0 + KC) * D_DIM;
            const float* nv = Vg + (size_t)(kb0 + KC) * D_DIM;
            __builtin_prefetch(nk + (size_t)tid * 64, 0, 1);
            __builtin_prefetch(nv + (size_t)tid * 64, 0, 1);
        }
        __syncthreads();   // K/V staged before use

        // ---- scores: S[16x16] = Q(16xD) * K_tile^T, full-D reduction ----
        v8f c = (v8f){0.f,0.f,0.f,0.f,0.f,0.f,0.f,0.f};
        #pragma unroll
        for (int d0 = 0; d0 < D_DIM; d0 += 32) {
            v16h a  = lds_loadA(Qs + wm * 16 * QP, QP, d0, lane);
            v16h bm = lds_loadB(Ks + wk * 16 * KP, KP, d0, lane);
            c = wmma_f16(a, bm, c);
        }
        {
            int n    = lane & 15;
            int keyg = kb0 + wk * 16 + n;
            #pragma unroll
            for (int r = 0; r < 8; ++r) {
                int ml = wm * 16 + hi * 8 + r;    // local query row
                int qg = qbase + ml;
                float dist = (float)(qg - keyg);
                float s = c[r] * scale + 1.0f / (1.0f + 0.2f * dist * dist);
                s = (keyg > qg) ? -3.0e38f : s;   // causal mask
                Sc[ml * SCP + wk * 16 + n] = s;
            }
        }
        __syncthreads();

        // ---- online softmax over this 64-key chunk (8 threads per row) ----
        {
            int row = tid >> 3;            // 0..63
            int cg  = (tid & 7) * 8;
            float sv[8];
            #pragma unroll
            for (int i = 0; i < 8; ++i) sv[i] = Sc[row * SCP + cg + i];
            float mx = sv[0];
            #pragma unroll
            for (int i = 1; i < 8; ++i) mx = fmaxf(mx, sv[i]);
            #pragma unroll
            for (int off = 1; off < 8; off <<= 1)
                mx = fmaxf(mx, __shfl_xor(mx, off, 32));
            float mo = smM[row];
            float mn = fmaxf(mo, mx);
            float al = __expf(mo - mn);
            float ps = 0.0f;
            #pragma unroll
            for (int i = 0; i < 8; ++i) {
                float p = __expf(sv[i] - mn);
                Pb[row * PP + cg + i] = (_Float16)p;
                ps += p;
            }
            #pragma unroll
            for (int off = 1; off < 8; off <<= 1)
                ps += __shfl_xor(ps, off, 32);
            if ((tid & 7) == 0) {
                smM[row] = mn;
                smL[row] = smL[row] * al + ps;
                smA[row] = al;
            }
        }
        __syncthreads();

        // ---- O += P * V : each wave owns 16 rows x 128 cols ----
        {
            #pragma unroll
            for (int r = 0; r < 8; ++r) {
                float al = smA[wm * 16 + hi * 8 + r];
                #pragma unroll
                for (int t = 0; t < 8; ++t) acc[t][r] *= al;
            }
            #pragma unroll
            for (int kk = 0; kk < KC; kk += 32) {
                v16h a = lds_loadA(Pb + wm * 16 * PP, PP, kk, lane);
                #pragma unroll
                for (int t = 0; t < 8; ++t) {
                    v16h bm = lds_loadB(Vt + (wd * 128 + t * 16) * VTP, VTP, kk, lane);
                    acc[t] = wmma_f16(a, bm, acc[t]);
                }
            }
        }
        __syncthreads();   // protect Ks/Vt/Sc/Pb before next chunk overwrites
    }

    // ---- epilogue: normalize by l and store ----
    {
        int n = lane & 15;
        float* Orow = O + ((size_t)bidx * S_LEN + qbase) * D_DIM;
        #pragma unroll
        for (int r = 0; r < 8; ++r) {
            int ml = wm * 16 + hi * 8 + r;
            float linv = 1.0f / smL[ml];
            #pragma unroll
            for (int t = 0; t < 8; ++t)
                Orow[(size_t)ml * D_DIM + wd * 128 + t * 16 + n] = acc[t][r] * linv;
        }
    }
}

extern "C" void kernel_launch(void* const* d_in, const int* in_sizes, int n_in,
                              void* d_out, int out_size, void* d_ws, size_t ws_size,
                              hipStream_t stream) {
    (void)in_sizes; (void)n_in; (void)out_size; (void)d_ws; (void)ws_size;
    const float* Q = (const float*)d_in[0];
    const float* K = (const float*)d_in[1];
    const float* V = (const float*)d_in[2];
    float* O = (float*)d_out;

    // allow large dynamic LDS (idempotent, not a stream op; capture-safe)
    (void)hipFuncSetAttribute((const void*)flash_attn_cauchy_kernel,
                              hipFuncAttributeMaxDynamicSharedMemorySize,
                              SMEM_BYTES);

    dim3 grid(S_LEN / MQ, BATCH);
    dim3 block(NTHREADS);
    flash_attn_cauchy_kernel<<<grid, block, SMEM_BYTES, stream>>>(Q, K, V, O);
}
